// MultiHeadedAttention_4492535792141
// MI455X (gfx1250) — compile-verified, matrix operand order checked by reading, not run on hardware
//
#include <hip/hip_runtime.h>
#include <hip/hip_bf16.h>

// ---------------------------------------------------------------------------
// MHA for MI455X (gfx1250): bf16 WMMA pipeline with async global->LDS staging
//   k_cvt      : fp32 -> bf16 elementwise, float4 vectorized
//   k_cvt_t    : fp32 [K,N] -> bf16 [N,K] (weights, LDS tiled transpose)
//   k_gemm     : C = A(bf16) @ Wt(bf16)^T + bias, 128x128 tile, 8 waves,
//                v_wmma_f32_16x16x32_bf16, double-buffered async-LDS staging
//   k_attn     : flash attention per (b,h,q-tile), online softmax,
//                double-buffered K/V tiles, scores + P@V via WMMA
// ---------------------------------------------------------------------------

typedef __attribute__((ext_vector_type(16))) __bf16 v16bf;
typedef __attribute__((ext_vector_type(8)))  float  v8f;

#define B_   4
#define H_   16
#define S_   2048
#define D_   1024
#define DK   64
#define MTOT (B_ * S_)   // 8192

__device__ __forceinline__ v8f wmma_bf16(v16bf a, v16bf b, v8f c) {
  // D = A(16x32) * B(32x16) + C, f32 accum
  return __builtin_amdgcn_wmma_f32_16x16x32_bf16(
      /*neg_a=*/false, a, /*neg_b=*/false, b,
      /*c_mod=*/(short)0, c, /*reuse_a=*/false, /*reuse_b=*/false);
}

union Frag32B { v16bf v; uint4 q[2]; };

__device__ __forceinline__ v16bf ld_frag(const __bf16* p) {
  Frag32B f;
  f.q[0] = *(const uint4*)(p);
  f.q[1] = *(const uint4*)(p + 8);
  return f.v;
}

// CDNA5 async copy: 16B global -> LDS, tracked by ASYNCcnt (no VGPR staging).
// LDS operand is the AS3 byte offset = low 32 bits of the generic pointer.
__device__ __forceinline__ void async_g2l(__bf16* lds_ptr, const __bf16* g) {
  const unsigned lo = (unsigned)(unsigned long long)(void*)lds_ptr;
  asm volatile("global_load_async_to_lds_b128 %0, %1, off"
               :: "v"(lo), "v"(g) : "memory");
}

__device__ __forceinline__ void wait_async0() {
  asm volatile("s_wait_asynccnt 0" ::: "memory");
}

// ---------------------------------------------------------------------------
// fp32 -> bf16 convert, 4 elements / thread / iter
// ---------------------------------------------------------------------------
union BF4 { __bf16 h[4]; uint2 u; };

__global__ __launch_bounds__(256)
void k_cvt(const float* __restrict__ in, __bf16* __restrict__ out, int n4) {
  int i = blockIdx.x * 256 + threadIdx.x;
  const int stride = gridDim.x * 256;
  const float4* in4 = (const float4*)in;
  for (; i < n4; i += stride) {
    const float4 f = in4[i];
    BF4 b;
    b.h[0] = (__bf16)f.x; b.h[1] = (__bf16)f.y;
    b.h[2] = (__bf16)f.z; b.h[3] = (__bf16)f.w;
    *(uint2*)(out + (size_t)i * 4) = b.u;
  }
}

// fp32 W[K,N] -> bf16 Wt[N,K]
__global__ __launch_bounds__(256)
void k_cvt_t(const float* __restrict__ W, __bf16* __restrict__ Wt) {
  __shared__ float t[32][33];
  const int n0 = blockIdx.x * 32, k0 = blockIdx.y * 32;
  const int tx = threadIdx.x & 31, tg = threadIdx.x >> 5;  // 32 x 8
  for (int i = 0; i < 4; ++i) {
    int kk = tg + i * 8;
    t[kk][tx] = W[(size_t)(k0 + kk) * D_ + n0 + tx];
  }
  __syncthreads();
  for (int i = 0; i < 4; ++i) {
    int nn = tg + i * 8;
    Wt[(size_t)(n0 + nn) * D_ + k0 + tx] = (__bf16)t[tx][nn];
  }
}

// ---------------------------------------------------------------------------
// GEMM: out[M=8192, N=1024] = A[M,K=1024] @ Wt[N,K]^T + bias
// mode 0: bf16 out, layout [B,H,S,dk]      (Q, K projections)
// mode 1: bf16 out, layout [B,H,dk,S]      (V projection, pre-transposed)
// mode 2: fp32 out, layout [M,N] row-major (final projection)
// ---------------------------------------------------------------------------
__global__ __launch_bounds__(256)
void k_gemm(const __bf16* __restrict__ A, const __bf16* __restrict__ Bt,
            const float* __restrict__ bias,
            __bf16* __restrict__ outb, float* __restrict__ outf, int mode) {
  __shared__ __align__(16) __bf16 sA[2][128][40];   // 128 rows x 32 k (+pad)
  __shared__ __align__(16) __bf16 sB[2][128][40];   // 128 n    x 32 k (+pad)

  const int tid  = threadIdx.x;
  const int lane = tid & 31;
  const int wid  = tid >> 5;           // 8 waves
  const int wm   = wid & 3;            // wave M offset (x32)
  const int wn   = wid >> 2;           // wave N offset (x64)
  const int m0   = blockIdx.x * 128;
  const int n0   = blockIdx.y * 128;
  const int lr   = lane & 15;
  const int kh   = lane >> 4;          // k-half select for operand frags
  const int lrow = tid >> 1;           // tile staging: 2 threads per row
  const int lcol = (tid & 1) * 16;     // 16 bf16 (32B) each

  const v8f z = {0.f, 0.f, 0.f, 0.f, 0.f, 0.f, 0.f, 0.f};
  v8f acc[2][4];
  for (int i = 0; i < 2; ++i)
    for (int j = 0; j < 4; ++j) acc[i][j] = z;

  const __bf16* gA = A  + (size_t)(m0 + lrow) * D_ + lcol;
  const __bf16* gB = Bt + (size_t)(n0 + lrow) * D_ + lcol;

  auto load_tile = [&](int buf, int kt) {
    const __bf16* ga = gA + kt * 32;
    const __bf16* gb = gB + kt * 32;
    async_g2l(&sA[buf][lrow][lcol],     ga);
    async_g2l(&sA[buf][lrow][lcol + 8], ga + 8);
    async_g2l(&sB[buf][lrow][lcol],     gb);
    async_g2l(&sB[buf][lrow][lcol + 8], gb + 8);
    __builtin_prefetch(ga + 64, 0, 3);   // WGP-scope prefetch, kt+2
    __builtin_prefetch(gb + 64, 0, 3);
  };

  load_tile(0, 0);
  wait_async0();
  __syncthreads();

  for (int kt = 0; kt < D_ / 32; ++kt) {
    const int buf = kt & 1;
    if (kt + 1 < D_ / 32) load_tile(buf ^ 1, kt + 1);

    v16bf af[2], bfr[4];
    for (int i = 0; i < 2; ++i)
      af[i] = ld_frag(&sA[buf][wm * 32 + i * 16 + lr][kh * 16]);
    for (int j = 0; j < 4; ++j)
      bfr[j] = ld_frag(&sB[buf][wn * 64 + j * 16 + lr][kh * 16]);

    for (int i = 0; i < 2; ++i)
      for (int j = 0; j < 4; ++j)
        acc[i][j] = wmma_bf16(af[i], bfr[j], acc[i][j]);

    wait_async0();       // our async stores into buf^1 are done
    __syncthreads();     // everyone's are visible
  }

  // Epilogue. C layout: vgpr r, lanes 0-15 -> row r, lanes 16-31 -> row r+8.
  const int rbase = (lane >= 16) ? 8 : 0;
  for (int i = 0; i < 2; ++i)
    for (int j = 0; j < 4; ++j)
      for (int r = 0; r < 8; ++r) {
        const int m = m0 + wm * 32 + i * 16 + r + rbase;
        const int n = n0 + wn * 64 + j * 16 + lr;
        const float v = acc[i][j][r] + bias[n];
        if (mode == 2) {
          outf[(size_t)m * D_ + n] = v;
        } else {
          const int b = m >> 11, s = m & (S_ - 1);
          const int h = n >> 6,  d = n & 63;
          const size_t idx = (mode == 0)
              ? ((((size_t)(b * H_ + h)) * S_ + s) * DK + d)
              : ((((size_t)(b * H_ + h)) * DK + d) * S_ + s);
          outb[idx] = (__bf16)v;
        }
      }
}

// ---------------------------------------------------------------------------
// Flash attention. One workgroup per (b*h, q-tile of 128). 8 waves, each
// owning 16 q-rows. kv swept in 128-row tiles (double-buffered), online
// softmax in registers, both matmuls on WMMA.
// ---------------------------------------------------------------------------
__global__ __launch_bounds__(256)
void k_attn(const __bf16* __restrict__ Q,   // [B,H,S,dk]
            const __bf16* __restrict__ Kc,  // [B,H,S,dk]
            const __bf16* __restrict__ Vt,  // [B,H,dk,S]
            __bf16* __restrict__ X2) {      // [B*S, H*dk]
  __shared__ __align__(16) __bf16 sK[2][128][64];     // kv rows x dk
  __shared__ __align__(16) __bf16 sV[2][64][128];     // dk rows x kv
  __shared__ __align__(16) __bf16 sP[8][16][128];     // per-wave P transpose

  const int tid  = threadIdx.x;
  const int lane = tid & 31;
  const int wid  = tid >> 5;
  const int lr   = lane & 15;
  const int kh   = lane >> 4;
  const int rbase = (lane >= 16) ? 8 : 0;
  const int bh    = blockIdx.y;            // b*H + h
  const int q0row = blockIdx.x * 128;

  const __bf16* Qp = Q  + (size_t)bh * S_ * DK;
  const __bf16* Kp = Kc + (size_t)bh * S_ * DK;
  const __bf16* Vp = Vt + (size_t)bh * DK * S_;

  // Q fragments for this wave: rows [wid*16, wid*16+16), k = 0..63
  v16bf qf[2];
  {
    const __bf16* qr = Qp + (size_t)(q0row + wid * 16 + lr) * DK;
    qf[0] = ld_frag(qr + kh * 16);        // k 0..31
    qf[1] = ld_frag(qr + 32 + kh * 16);   // k 32..63
  }

  const v8f z = {0.f, 0.f, 0.f, 0.f, 0.f, 0.f, 0.f, 0.f};
  float rmax[8], rsum[8];
  v8f oacc[4];
  for (int r = 0; r < 8; ++r) { rmax[r] = -1e30f; rsum[r] = 0.f; }
  for (int d = 0; d < 4; ++d) oacc[d] = z;

  const int krow = tid >> 1, kc0 = (tid & 1) * 32;   // K staging: 2 thr/row
  const int vrow = tid >> 2, vc0 = (tid & 3) * 32;   // V staging: 4 thr/row

  auto load_kv = [&](int buf, int kt) {
    const int kv0 = kt * 128;
    const __bf16* gk = Kp + (size_t)(kv0 + krow) * DK + kc0;
    async_g2l(&sK[buf][krow][kc0],      gk);
    async_g2l(&sK[buf][krow][kc0 + 8],  gk + 8);
    async_g2l(&sK[buf][krow][kc0 + 16], gk + 16);
    async_g2l(&sK[buf][krow][kc0 + 24], gk + 24);
    const __bf16* gv = Vp + (size_t)vrow * S_ + kv0 + vc0;
    async_g2l(&sV[buf][vrow][vc0],      gv);
    async_g2l(&sV[buf][vrow][vc0 + 8],  gv + 8);
    async_g2l(&sV[buf][vrow][vc0 + 16], gv + 16);
    async_g2l(&sV[buf][vrow][vc0 + 24], gv + 24);
  };

  load_kv(0, 0);
  wait_async0();
  __syncthreads();

  for (int kt = 0; kt < S_ / 128; ++kt) {
    const int buf = kt & 1;
    if (kt + 1 < S_ / 128) load_kv(buf ^ 1, kt + 1);

    // scores S = (Q @ K^T) * 1/sqrt(dk) : 8 frags of 16x16, k-dim 64
    v8f sf[8];
    for (int j = 0; j < 8; ++j) {
      v16bf k0 = ld_frag(&sK[buf][j * 16 + lr][kh * 16]);
      v16bf k1 = ld_frag(&sK[buf][j * 16 + lr][32 + kh * 16]);
      v8f s = wmma_bf16(qf[0], k0, z);
      s = wmma_bf16(qf[1], k1, s);
      for (int r = 0; r < 8; ++r) s[r] = s[r] * 0.125f;  // 1/sqrt(64)
      sf[j] = s;
    }

    // online softmax: new row max (reduce across 8 frags + 16 lanes per half)
    float nm[8];
    for (int r = 0; r < 8; ++r) {
      float m = rmax[r];
      for (int j = 0; j < 8; ++j) m = fmaxf(m, sf[j][r]);
      for (int off = 8; off >= 1; off >>= 1)
        m = fmaxf(m, __shfl_xor(m, off));
      nm[r] = m;
    }
    for (int r = 0; r < 8; ++r) {
      const float sc = __expf(rmax[r] - nm[r]);
      rsum[r] = rsum[r] * sc;
      rmax[r] = nm[r];
      for (int d = 0; d < 4; ++d) oacc[d][r] = oacc[d][r] * sc;
    }

    // P = exp(S - max): accumulate across j in registers, reduce once per r,
    // and transpose into A-operand layout through per-wave LDS.
    for (int r = 0; r < 8; ++r) {
      float a = 0.f;
      for (int j = 0; j < 8; ++j) {
        const float p = __expf(sf[j][r] - nm[r]);
        a += p;
        sP[wid][r + rbase][j * 16 + lr] = (__bf16)p;
      }
      for (int off = 8; off >= 1; off >>= 1)
        a += __shfl_xor(a, off);
      rsum[r] += a;
    }
    // in-wave LDS RAW: wait all ds stores before transposed reads
    asm volatile("s_wait_dscnt 0" ::: "memory");
    __builtin_amdgcn_wave_barrier();

    // O += P(16x128) @ V(128x64) as 4 k-chunks x 4 d-frags
    for (int kk = 0; kk < 4; ++kk) {
      v16bf ap = ld_frag(&sP[wid][lr][kk * 32 + kh * 16]);
      for (int d = 0; d < 4; ++d) {
        v16bf bv = ld_frag(&sV[buf][d * 16 + lr][kk * 32 + kh * 16]);
        oacc[d] = wmma_bf16(ap, bv, oacc[d]);
      }
    }

    wait_async0();       // next tile's async stores (buf^1) complete
    __syncthreads();     // visible to all waves before next iteration
  }

  // normalize and write merged-head output [B*S, H*dk]
  const int b = bh >> 4, h = bh & 15;
  for (int d = 0; d < 4; ++d)
    for (int r = 0; r < 8; ++r) {
      const int srow = q0row + wid * 16 + r + rbase;
      const int col  = h * 64 + d * 16 + lr;
      const float v = oacc[d][r] / rsum[r];
      X2[((size_t)(b * S_ + srow)) * D_ + col] = (__bf16)v;
    }
}

// ---------------------------------------------------------------------------
extern "C" void kernel_launch(void* const* d_in, const int* in_sizes, int n_in,
                              void* d_out, int out_size, void* d_ws, size_t ws_size,
                              hipStream_t stream) {
  const float* q  = (const float*)d_in[0];
  const float* k  = (const float*)d_in[1];
  const float* v  = (const float*)d_in[2];
  const float* Wq = (const float*)d_in[3];
  const float* bq = (const float*)d_in[4];
  const float* Wk = (const float*)d_in[5];
  const float* bk = (const float*)d_in[6];
  const float* Wv = (const float*)d_in[7];
  const float* bv = (const float*)d_in[8];
  const float* Wo = (const float*)d_in[9];
  const float* bo = (const float*)d_in[10];
  float* out = (float*)d_out;

  const size_t NEI = (size_t)MTOT * D_;  // 8,388,608 activation elements
  const size_t NEW = (size_t)D_ * D_;    // 1,048,576 weight elements

  __bf16* p = (__bf16*)d_ws;
  __bf16* qin = p; p += NEI;
  __bf16* kin = p; p += NEI;
  __bf16* vin = p; p += NEI;
  __bf16* Wqt = p; p += NEW;
  __bf16* Wkt = p; p += NEW;
  __bf16* Wvt = p; p += NEW;
  __bf16* Wot = p; p += NEW;
  __bf16* Qh  = p; p += NEI;   // [B,H,S,dk]
  __bf16* Kh  = p; p += NEI;   // [B,H,S,dk]
  __bf16* Vth = p; p += NEI;   // [B,H,dk,S]
  __bf16* X2  = p; p += NEI;   // [B*S, H*dk]

  const dim3 blk(256);
  const int n4 = (int)(NEI / 4);

  k_cvt<<<dim3(2048), blk, 0, stream>>>(q, qin, n4);
  k_cvt<<<dim3(2048), blk, 0, stream>>>(k, kin, n4);
  k_cvt<<<dim3(2048), blk, 0, stream>>>(v, vin, n4);

  k_cvt_t<<<dim3(32, 32), blk, 0, stream>>>(Wq, Wqt);
  k_cvt_t<<<dim3(32, 32), blk, 0, stream>>>(Wk, Wkt);
  k_cvt_t<<<dim3(32, 32), blk, 0, stream>>>(Wv, Wvt);
  k_cvt_t<<<dim3(32, 32), blk, 0, stream>>>(Wo, Wot);

  const dim3 ggrid(MTOT / 128, D_ / 128);
  k_gemm<<<ggrid, blk, 0, stream>>>(qin, Wqt, bq, Qh,  nullptr, 0);
  k_gemm<<<ggrid, blk, 0, stream>>>(kin, Wkt, bk, Kh,  nullptr, 0);
  k_gemm<<<ggrid, blk, 0, stream>>>(vin, Wvt, bv, Vth, nullptr, 1);

  k_attn<<<dim3(S_ / 128, B_ * H_), blk, 0, stream>>>(Qh, Kh, Vth, X2);

  k_gemm<<<ggrid, blk, 0, stream>>>(X2, Wot, bo, nullptr, out, 2);
}
